// RNN_36421322670148
// MI455X (gfx1250) — matrix operand order repaired and checked by reading, not compiled, not run on hardware
//
#include <hip/hip_runtime.h>
#include <hip/hip_bf16.h>
#include <cstdint>

// ---------------------------------------------------------------------------
// Problem constants (from the reference)
// ---------------------------------------------------------------------------
#define B_   256
#define T_   100
#define DIN_ 2
#define NG_  4096
#define NP_  512

typedef __bf16 bf16;
typedef __attribute__((ext_vector_type(16))) __bf16 v16bf;
typedef __attribute__((ext_vector_type(8)))  __bf16 v8bf;
typedef __attribute__((ext_vector_type(8)))  float  v8f;

// ---------------------------------------------------------------------------
// Helpers
// ---------------------------------------------------------------------------
__device__ __forceinline__ v16bf cat8(v8bf a, v8bf b) {
  return __builtin_shufflevector(a, b, 0, 1, 2, 3, 4, 5, 6, 7,
                                       8, 9, 10, 11, 12, 13, 14, 15);
}

__device__ __forceinline__ v8f wmma_bf16(v16bf a, v16bf b, v8f c) {
  // D(f32 16x16) = A(16x32 bf16) * B(32x16 bf16) + C
  return __builtin_amdgcn_wmma_f32_16x16x32_bf16(
      /*neg_a=*/false, a, /*neg_b=*/false, b,
      /*c_mod=*/(short)0, c, /*reuse_a=*/false, /*reuse_b=*/false);
}

// One async 16B-per-lane copy: LDS[ldsOff] = MEM[ga..ga+15], tracked by ASYNCcnt
__device__ __forceinline__ void async_b128(unsigned ldsOff, uint64_t ga) {
  asm volatile("global_load_async_to_lds_b128 %0, %1, off"
               :: "v"(ldsOff), "v"(ga)
               : "memory");
}

template <int N>
__device__ __forceinline__ void wait_async() {
#if __has_builtin(__builtin_amdgcn_s_wait_asynccnt)
  __builtin_amdgcn_s_wait_asynccnt((short)N);
#else
  asm volatile("s_wait_asynccnt %0" :: "n"(N) : "memory");
#endif
}

// ---------------------------------------------------------------------------
// Split fp32 -> (hi, lo) bf16   (Markidis decomposition, exact to ~2^-16)
// ---------------------------------------------------------------------------
__global__ void split_f32_bf16x2(const float* __restrict__ src,
                                 bf16* __restrict__ hi,
                                 bf16* __restrict__ lo, int n) {
  int i = blockIdx.x * blockDim.x + threadIdx.x;
  int stride = gridDim.x * blockDim.x;
  for (; i < n; i += stride) {
    float f = src[i];
    bf16 h = (bf16)f;
    hi[i] = h;
    lo[i] = (bf16)(f - (float)h);
  }
}

// ---------------------------------------------------------------------------
// Split-bf16 WMMA GEMM:  C[M x N] = A[M x K] * B[K x N]   (row-major all)
//   3 x v_wmma_f32_16x16x32_bf16 per K=32 chunk (hi*hi + hi*lo + lo*hi),
//   f32 accumulation.  Each wave owns a 32-row x BN tile (2 M-stripes) so
//   every B fragment read from LDS feeds 6 WMMAs (ds:wmma = 1:1).
//   Double-buffered LDS stages filled by GLOBAL_LOAD_ASYNC_TO_LDS_B128,
//   pipelined on ASYNCcnt.
// Epilogue modes:
//   MODE_ENC : h  = acc                           -> store hi/lo bf16
//   MODE_STEP: h' = relu(acc + v0*Win0 + v1*Win1) -> store hi/lo bf16
//   MODE_DEC : out= acc                           -> store fp32
// ---------------------------------------------------------------------------
enum { MODE_ENC = 0, MODE_STEP = 1, MODE_DEC = 2 };

template <int BM, int BN, int MODE>
__global__ __launch_bounds__(BM) void wmma_gemm_split(
    const bf16* __restrict__ Ahi, const bf16* __restrict__ Alo, int lda,
    const bf16* __restrict__ Bhi, const bf16* __restrict__ Blo, int ldb,
    int K,
    bf16* __restrict__ Hhi, bf16* __restrict__ Hlo, int ldh,   // ENC / STEP
    const float* __restrict__ vbase, int ldv,                  // STEP: v + t*2
    const float* __restrict__ Win,                             // STEP: [2][NG]
    float* __restrict__ Cout, int ldc)                         // DEC
{
  constexpr int NT      = BN / 16;        // 16x16 tiles per wave (N dir)
  constexpr int NTHREAD = BM;             // 32 * (BM/32) : wave owns 32 rows
  constexpr int A_BYTES = BM * 32 * 2;    // one A tile (hi or lo)
  constexpr int B_BYTES = 32 * BN * 2;    // one B tile (hi or lo)
  constexpr int STAGE   = 2 * A_BYTES + 2 * B_BYTES;
  // per-wave async instructions issued per stage:
  constexpr int TRIPS_A  = (BM * 32) / (NTHREAD * 8);
  constexpr int TRIPS_B  = (32 * BN) / (NTHREAD * 8);
  constexpr int INFLIGHT = 2 * TRIPS_A + 2 * TRIPS_B;
  constexpr int XW_BYTES = (MODE == MODE_STEP) ? (2 * BM + 2 * BN) * 4 : 16;

  __shared__ __align__(16) char smem[2 * STAGE + XW_BYTES];

  const int tid  = threadIdx.x;
  const int lane = tid & 31;
  const int wave = tid >> 5;
  const int m0   = blockIdx.y * BM;
  const int n0   = blockIdx.x * BN;

  float* sx0 = (float*)(smem + 2 * STAGE);          // [BM]
  float* sx1 = sx0 + BM;                            // [BM]
  float* sw0 = sx1 + BM;                            // [BN]
  float* sw1 = sw0 + BN;                            // [BN]

  if constexpr (MODE == MODE_STEP) {
    // fused input projection: x[b,g] = v[b,0]*Win[0,g] + v[b,1]*Win[1,g]
    for (int r = tid; r < BM; r += NTHREAD) {
      sx0[r] = vbase[(size_t)(m0 + r) * ldv + 0];
      sx1[r] = vbase[(size_t)(m0 + r) * ldv + 1];
    }
    for (int c = tid; c < BN; c += NTHREAD) {
      sw0[c] = Win[n0 + c];
      sw1[c] = Win[NG_ + n0 + c];
    }
  }

  // Issue all async copies for one stage (buf in {0,1}) at K offset kk.
  auto issue_stage = [&](int buf, int kk) {
    const unsigned sbase = buf * STAGE;
#pragma unroll
    for (int i = 0; i < TRIPS_A; ++i) {
      const int idx = (tid + i * NTHREAD) * 8;      // bf16 elements
      const int row = idx >> 5, kc = idx & 31;
      const size_t g = (size_t)(m0 + row) * lda + kk + kc;
      async_b128(sbase + idx * 2,            (uint64_t)&Ahi[g]);
      async_b128(sbase + A_BYTES + idx * 2,  (uint64_t)&Alo[g]);
    }
#pragma unroll
    for (int i = 0; i < TRIPS_B; ++i) {
      const int idx = (tid + i * NTHREAD) * 8;
      const int kr = idx / BN, nc = idx % BN;
      const size_t g = (size_t)(kk + kr) * ldb + n0 + nc;
      async_b128(sbase + 2 * A_BYTES + idx * 2,            (uint64_t)&Bhi[g]);
      async_b128(sbase + 2 * A_BYTES + B_BYTES + idx * 2,  (uint64_t)&Blo[g]);
    }
  };

  v8f acc[2][NT];
#pragma unroll
  for (int s = 0; s < 2; ++s)
#pragma unroll
    for (int i = 0; i < NT; ++i) acc[s][i] = (v8f)0.0f;

  issue_stage(0, 0);                                  // prologue prefetch

  int buf = 0;
  for (int kk = 0; kk < K; kk += 32) {
    // overlap: start next slab's L2->LDS DMA before computing current slab
    if (kk + 32 < K) {
      issue_stage(buf ^ 1, kk + 32);
      wait_async<INFLIGHT>();   // ASYNCcnt in-order => current stage landed
    } else {
      wait_async<0>();
    }
    __syncthreads();            // all waves' stage done

    const bf16* sA_hi = (const bf16*)(smem + buf * STAGE);
    const bf16* sA_lo = sA_hi + BM * 32;
    const bf16* sB_hi = sA_lo + BM * 32;
    const bf16* sB_lo = sB_hi + 32 * BN;

    // ---- A fragments: 2 M-stripes per wave (ISA 16-bit A 16x32 layout) ----
    const int ak0 = (lane >= 16) ? 8 : 0;   // lanes>=16 hold K=8..15,24..31
    v16bf a_hi[2], a_lo[2];
#pragma unroll
    for (int s = 0; s < 2; ++s) {
      const int arow = wave * 32 + s * 16 + (lane & 15);
      a_hi[s] = cat8(*(const v8bf*)&sA_hi[arow * 32 + ak0],
                     *(const v8bf*)&sA_hi[arow * 32 + ak0 + 16]);
      a_lo[s] = cat8(*(const v8bf*)&sA_lo[arow * 32 + ak0],
                     *(const v8bf*)&sA_lo[arow * 32 + ak0 + 16]);
    }

#pragma unroll
    for (int nt = 0; nt < NT; ++nt) {
      // ---- B fragment: lane&15 = N column, K in same order as A (ak0.., ak0+16..) ----
      const int bn = nt * 16 + (lane & 15);
      v16bf b_hi, b_lo;
#pragma unroll
      for (int e = 0; e < 16; ++e) {
        const int bk = ak0 + (e < 8 ? e : e + 8);
        b_hi[e] = sB_hi[bk * BN + bn];
        b_lo[e] = sB_lo[bk * BN + bn];
      }
#pragma unroll
      for (int s = 0; s < 2; ++s) {       // each B fragment feeds 6 WMMAs
        acc[s][nt] = wmma_bf16(a_hi[s], b_hi, acc[s][nt]);
        acc[s][nt] = wmma_bf16(a_hi[s], b_lo, acc[s][nt]);
        acc[s][nt] = wmma_bf16(a_lo[s], b_hi, acc[s][nt]);
      }
    }
    __syncthreads();            // everyone done reading buf; safe to overwrite
    buf ^= 1;
  }

  // ---- epilogue (C 16x16 f32 layout: VGPR r -> M=r / r+8, N = lane&15) ----
  const int nlo    = lane & 15;
  const int rowOff = (lane >= 16) ? 8 : 0;
#pragma unroll
  for (int s = 0; s < 2; ++s) {
#pragma unroll
    for (int nt = 0; nt < NT; ++nt) {
      const int colL = nt * 16 + nlo;
      const int col  = n0 + colL;
#pragma unroll
      for (int r = 0; r < 8; ++r) {
        const int rowL = wave * 32 + s * 16 + rowOff + r;
        const int row  = m0 + rowL;
        float val = acc[s][nt][r];
        if constexpr (MODE == MODE_STEP) {
          val += sx0[rowL] * sw0[colL] + sx1[rowL] * sw1[colL];
          val = fmaxf(val, 0.0f);
        }
        if constexpr (MODE == MODE_DEC) {
          Cout[(size_t)row * ldc + col] = val;
        } else {
          bf16 h = (bf16)val;
          Hhi[(size_t)row * ldh + col] = h;
          Hlo[(size_t)row * ldh + col] = (bf16)(val - (float)h);
        }
      }
    }
  }
}

// ---------------------------------------------------------------------------
// Host-side orchestration
// ---------------------------------------------------------------------------
extern "C" void kernel_launch(void* const* d_in, const int* in_sizes, int n_in,
                              void* d_out, int out_size, void* d_ws,
                              size_t ws_size, hipStream_t stream) {
  (void)in_sizes; (void)n_in; (void)out_size; (void)ws_size;

  const float* v    = (const float*)d_in[0];  // [B, T, 2]
  const float* P0   = (const float*)d_in[1];  // [B, NP]
  const float* Wenc = (const float*)d_in[2];  // [NP, NG]
  const float* Win  = (const float*)d_in[3];  // [2, NG]
  const float* Wrec = (const float*)d_in[4];  // [NG, NG]
  const float* Wdec = (const float*)d_in[5];  // [NG, NP]
  float* out = (float*)d_out;                 // [B, T, NP]

  // workspace carve-out (256B aligned)
  size_t off = 0;
  auto carve = [&](size_t bytes) -> char* {
    char* p = (char*)d_ws + off;
    off += (bytes + 255) & ~(size_t)255;
    return p;
  };
  bf16* Wrec_hi = (bf16*)carve((size_t)NG_ * NG_ * 2);
  bf16* Wrec_lo = (bf16*)carve((size_t)NG_ * NG_ * 2);
  bf16* Wenc_hi = (bf16*)carve((size_t)NP_ * NG_ * 2);
  bf16* Wenc_lo = (bf16*)carve((size_t)NP_ * NG_ * 2);
  bf16* Wdec_hi = (bf16*)carve((size_t)NG_ * NP_ * 2);
  bf16* Wdec_lo = (bf16*)carve((size_t)NG_ * NP_ * 2);
  bf16* P0_hi   = (bf16*)carve((size_t)B_ * NP_ * 2);
  bf16* P0_lo   = (bf16*)carve((size_t)B_ * NP_ * 2);
  bf16* h_hi[2] = {(bf16*)carve((size_t)B_ * NG_ * 2),
                   (bf16*)carve((size_t)B_ * NG_ * 2)};
  bf16* h_lo[2] = {(bf16*)carve((size_t)B_ * NG_ * 2),
                   (bf16*)carve((size_t)B_ * NG_ * 2)};

  // 1) split weights / P0 into hi+lo bf16
  auto split = [&](const float* s, bf16* hi, bf16* lo, int n) {
    int blocks = (n + 256 * 8 - 1) / (256 * 8);
    if (blocks > 2048) blocks = 2048;
    split_f32_bf16x2<<<blocks, 256, 0, stream>>>(s, hi, lo, n);
  };
  split(Wrec, Wrec_hi, Wrec_lo, NG_ * NG_);
  split(Wenc, Wenc_hi, Wenc_lo, NP_ * NG_);
  split(Wdec, Wdec_hi, Wdec_lo, NG_ * NP_);
  split(P0, P0_hi, P0_lo, B_ * NP_);

  // 2) encoder: h0 = P0 @ W_enc   [256 x 4096], K = 512
  wmma_gemm_split<128, 64, MODE_ENC>
      <<<dim3(NG_ / 64, B_ / 128), 128, 0, stream>>>(
          P0_hi, P0_lo, NP_, Wenc_hi, Wenc_lo, NG_, NP_,
          h_hi[0], h_lo[0], NG_,
          nullptr, 0, nullptr, nullptr, 0);

  // 3) recurrence + per-step decode
  for (int t = 0; t < T_; ++t) {
    const int cur = t & 1, nxt = cur ^ 1;
    // h_{t+1} = relu(v[:,t,:] @ W_in + h_t @ W_rec)   [256 x 4096], K = 4096
    wmma_gemm_split<128, 64, MODE_STEP>
        <<<dim3(NG_ / 64, B_ / 128), 128, 0, stream>>>(
            h_hi[cur], h_lo[cur], NG_, Wrec_hi, Wrec_lo, NG_, NG_,
            h_hi[nxt], h_lo[nxt], NG_,
            v + (size_t)t * DIN_, T_ * DIN_, Win,
            nullptr, 0);
    // out[:, t, :] = h_{t+1} @ W_dec   [256 x 512], K = 4096
    wmma_gemm_split<64, 64, MODE_DEC>
        <<<dim3(NP_ / 64, B_ / 64), 64, 0, stream>>>(
            h_hi[nxt], h_lo[nxt], NG_, Wdec_hi, Wdec_lo, NP_, NG_,
            nullptr, nullptr, 0,
            nullptr, 0, nullptr,
            out + (size_t)t * NP_, T_ * NP_);
  }
}
